// VQVAE_Encoder_57269093925121
// MI455X (gfx1250) — matrix operand order repaired and checked by reading, not compiled, not run on hardware
//
#include <hip/hip_runtime.h>

// ---------------------------------------------------------------------------
// CDNA5 (gfx1250, wave32) types & helpers
// ---------------------------------------------------------------------------
typedef __bf16 v16bf __attribute__((ext_vector_type(16)));
typedef float  v8f   __attribute__((ext_vector_type(8)));

union FragU { unsigned int u[8]; v16bf v; };

__device__ __forceinline__ unsigned short f2bf(float x) {  // RNE fp32 -> bf16
  unsigned int u = __float_as_uint(x);
  return (unsigned short)((u + 0x7FFFu + ((u >> 16) & 1u)) >> 16);
}
__device__ __forceinline__ float bf2f(unsigned short h) {
  return __uint_as_float(((unsigned int)h) << 16);
}

// LDS byte offset of a generic pointer to __shared__ memory
__device__ __forceinline__ unsigned lds_off(const void* p) {
  return (unsigned)(uintptr_t)(__attribute__((address_space(3))) const void*)p;
}
// CDNA5 async global->LDS copy (ASYNCcnt-tracked); per-lane LDS destination.
__device__ __forceinline__ void async_load_b128(unsigned lds_addr, const void* gptr) {
  asm volatile("global_load_async_to_lds_b128 %0, %1, off"
               :: "v"(lds_addr), "v"(gptr) : "memory");
}
__device__ __forceinline__ void wait_async0() {
  asm volatile("s_wait_asynccnt 0x0" ::: "memory");
}

// ---------------------------------------------------------------------------
// NT bf16 WMMA GEMM:  C[M,N] = act( A[M,K] @ Bt[N,K]^T + bias )
//   A = [A0 | A1] concatenated along K at split K0 (A1 unused if K0 == K).
//   OUTBF: 1 -> bf16 (ushort) output, 0 -> fp32 output. ACT==1 -> LeakyReLU.
// Block 256 threads = 8 waves; tile 128x128; wave tile 32x64 (2x4 WMMA).
// Tiles staged to LDS with async b128 copies, double-buffered; rows padded
// to 80B (20 dwords) so 16-lane fragment gathers are bank-conflict free.
// M,N multiples of 128; K multiple of 32.
// ---------------------------------------------------------------------------
template<int ACT, int OUTBF>
__global__ __launch_bounds__(256)
void gemm_nt_bf16(const unsigned short* __restrict__ A0,
                  const unsigned short* __restrict__ A1, int K0,
                  const unsigned short* __restrict__ Bt,
                  const float* __restrict__ bias, void* __restrict__ Cv,
                  int M, int N, int K)
{
  __shared__ unsigned int tiles[2][2][128 * 20];  // [buf][A=0/B=1][row*20+dw]

  const int t    = threadIdx.x;
  const int lane = t & 31;
  const int wave = t >> 5;
  const int wm   = wave & 3;   // M offset 32*wm
  const int wn   = wave >> 2;  // N offset 64*wn
  const int m0   = blockIdx.y * 128;
  const int n0   = blockIdx.x * 128;
  const int hl   = lane >> 4;
  const int lm   = lane & 15;
  const int KA1  = K - K0;

  v8f acc[2][4];
  #pragma unroll
  for (int i = 0; i < 2; i++)
    #pragma unroll
    for (int j = 0; j < 4; j++)
      #pragma unroll
      for (int r = 0; r < 8; r++) acc[i][j][r] = 0.0f;

  // Issue async copies of one 128x32 A tile and one 128x32 B tile.
  // 512 16-byte chunks per tile; each thread issues 2 A + 2 B chunks.
  auto issue_tile = [&](int buf, int kt) {
    #pragma unroll
    for (int i = 0; i < 2; i++) {
      int c = i * 256 + t;               // 0..511
      int row = c >> 2, seg = c & 3;     // seg: 16B (8 bf16) within 64B row
      int gk = kt + seg * 8;
      int gm = m0 + row;
      const unsigned short* src = (gk < K0)
          ? (A0 + (size_t)gm * K0 + gk)
          : (A1 + (size_t)gm * KA1 + (gk - K0));
      async_load_b128(lds_off(&tiles[buf][0][row * 20 + seg * 4]), src);
    }
    #pragma unroll
    for (int i = 0; i < 2; i++) {
      int c = i * 256 + t;
      int row = c >> 2, seg = c & 3;
      const unsigned short* src = Bt + (size_t)(n0 + row) * K + kt + seg * 8;
      async_load_b128(lds_off(&tiles[buf][1][row * 20 + seg * 4]), src);
    }
  };

  issue_tile(0, 0);
  wait_async0();
  __syncthreads();

  const int nk = K >> 5;
  for (int it = 0; it < nk; it++) {
    const int cur = it & 1;
    if (it + 1 < nk) issue_tile(cur ^ 1, (it + 1) << 5);  // overlap with WMMA

    const unsigned int* At = &tiles[cur][0][0];
    const unsigned int* Bl = &tiles[cur][1][0];

    // B fragments: lane holds N=l%16, K = 16*(l/16) + 2r..2r+1
    FragU fb[4];
    #pragma unroll
    for (int nt = 0; nt < 4; nt++) {
      int n = wn * 64 + nt * 16 + lm;
      #pragma unroll
      for (int r = 0; r < 8; r++) fb[nt].u[r] = Bl[n * 20 + hl * 8 + r];
    }
    // A fragments + WMMA
    #pragma unroll
    for (int mt = 0; mt < 2; mt++) {
      FragU fa;
      int row = wm * 32 + mt * 16 + lm;
      #pragma unroll
      for (int r = 0; r < 4; r++) fa.u[r]     = At[row * 20 + hl * 8 + r];
      #pragma unroll
      for (int r = 0; r < 4; r++) fa.u[4 + r] = At[row * 20 + hl * 8 + 4 + r];
      #pragma unroll
      for (int nt = 0; nt < 4; nt++) {
        acc[mt][nt] = __builtin_amdgcn_wmma_f32_16x16x32_bf16(
            false, fa.v, false, fb[nt].v, (short)0, acc[mt][nt], false, false);
      }
    }
    wait_async0();
    __syncthreads();
  }

  // epilogue: m = base + r + 8*(lane/16), n = base + lane%16
  #pragma unroll
  for (int mt = 0; mt < 2; mt++) {
    #pragma unroll
    for (int nt = 0; nt < 4; nt++) {
      int n = n0 + wn * 64 + nt * 16 + lm;
      float bv = bias ? bias[n] : 0.0f;
      #pragma unroll
      for (int r = 0; r < 8; r++) {
        int m = m0 + wm * 32 + mt * 16 + hl * 8 + r;
        float vv = acc[mt][nt][r] + bv;
        if (ACT == 1) vv = vv > 0.0f ? vv : 0.2f * vv;
        if (OUTBF) ((unsigned short*)Cv)[(size_t)m * N + n] = f2bf(vv);
        else       ((float*)Cv)[(size_t)m * N + n] = vv;
      }
    }
  }
}

// ---------------------------------------------------------------------------
// Transpose-convert: fp32 src[K,N] -> bf16 dst[N,K]
// ---------------------------------------------------------------------------
__global__ __launch_bounds__(256)
void transpose_convert(const float* __restrict__ src,
                       unsigned short* __restrict__ dst, int K, int N)
{
  __shared__ float tile[32][33];
  const int t  = threadIdx.x;
  const int tx = t & 31, ty = t >> 5;     // ty 0..7
  const int k0 = blockIdx.y * 32, n0 = blockIdx.x * 32;
  #pragma unroll
  for (int j = 0; j < 4; j++)
    tile[ty + 8 * j][tx] = src[(size_t)(k0 + ty + 8 * j) * N + n0 + tx];
  __syncthreads();
  #pragma unroll
  for (int j = 0; j < 4; j++)
    dst[(size_t)(n0 + ty + 8 * j) * K + k0 + tx] = f2bf(tile[tx][ty + 8 * j]);
}

// Flat fp32 -> bf16 convert (count multiple of 256)
__global__ __launch_bounds__(256)
void convert_bf16(const float* __restrict__ src, unsigned short* __restrict__ dst)
{
  size_t i = (size_t)blockIdx.x * 256 + threadIdx.x;
  dst[i] = f2bf(src[i]);
}

// ---------------------------------------------------------------------------
// Per-segment attention + mean pool (bf16 in, bf16 out); one block per segment.
// pooled[b] = sum_m (mean_l attn[l,m]) * v[m]   (mean commutes with attn@V)
// ---------------------------------------------------------------------------
__global__ __launch_bounds__(64)
void attn_pool(const unsigned short* __restrict__ q,
               const unsigned short* __restrict__ k,
               const unsigned short* __restrict__ v,
               unsigned short* __restrict__ pooled)
{
  __shared__ float sattn[8][8];
  __shared__ float sw[8];
  const int b = blockIdx.x;
  const int t = threadIdx.x;
  const int l = t >> 3, m = t & 7;

  const unsigned short* qrow = q + (size_t)b * 8192 + l * 1024;
  const unsigned short* krow = k + (size_t)b * 8192 + m * 1024;
  float s = 0.0f;
  #pragma unroll 8
  for (int e = 0; e < 1024; e++) s += bf2f(qrow[e]) * bf2f(krow[e]);
  sattn[l][m] = s * (1.0f / 32.0f);   // / sqrt(1024)
  __syncthreads();

  if (t < 8) {  // softmax over row t
    float mx = sattn[t][0];
    #pragma unroll
    for (int j = 1; j < 8; j++) mx = fmaxf(mx, sattn[t][j]);
    float ex[8], sum = 0.0f;
    #pragma unroll
    for (int j = 0; j < 8; j++) { ex[j] = __expf(sattn[t][j] - mx); sum += ex[j]; }
    float inv = 1.0f / sum;
    #pragma unroll
    for (int j = 0; j < 8; j++) sattn[t][j] = ex[j] * inv;
  }
  __syncthreads();
  if (t < 8) {  // column mean -> weights
    float ws_ = 0.0f;
    #pragma unroll
    for (int l2 = 0; l2 < 8; l2++) ws_ += sattn[l2][t];
    sw[t] = ws_ * 0.125f;
  }
  __syncthreads();

  #pragma unroll
  for (int j = 0; j < 16; j++) {
    int e = j * 64 + t;
    float acc = 0.0f;
    #pragma unroll
    for (int mm = 0; mm < 8; mm++)
      acc += sw[mm] * bf2f(v[(size_t)b * 8192 + mm * 1024 + e]);
    pooled[(size_t)b * 1024 + e] = f2bf(acc);
  }
}

// ---------------------------------------------------------------------------
// Codebook squared norms: en2[n] = sum_e E[n][e]^2  (fp32 codebook)
// ---------------------------------------------------------------------------
__global__ __launch_bounds__(256)
void enorm_kernel(const float* __restrict__ E, float* __restrict__ en2)
{
  __shared__ float red[256];
  const int n = blockIdx.x, t = threadIdx.x;
  float s = 0.0f;
  #pragma unroll
  for (int j = 0; j < 4; j++) {
    float x = E[(size_t)n * 1024 + j * 256 + t];
    s += x * x;
  }
  red[t] = s; __syncthreads();
  for (int off = 128; off > 0; off >>= 1) {
    if (t < off) red[t] += red[t + off];
    __syncthreads();
  }
  if (t == 0) en2[n] = red[0];
}

__global__ void init_kernel(float* __restrict__ mse) { *mse = 0.0f; }

// ---------------------------------------------------------------------------
// VQ: argmin_n(en2[n] - 2*S[b,n]) (first-min tie), gather, ST output, MSE.
// ---------------------------------------------------------------------------
__global__ __launch_bounds__(256)
void vq_kernel(const float* __restrict__ S, const float* __restrict__ en2,
               const float* __restrict__ xenc, const float* __restrict__ E,
               float* __restrict__ out_q, float* __restrict__ out_idx,
               float* __restrict__ mse_accum)
{
  __shared__ float rv[256];
  __shared__ int   ri[256];
  const int b = blockIdx.x, t = threadIdx.x;

  float best = 3.4e38f; int bidx = 0x7fffffff;
  #pragma unroll
  for (int j = 0; j < 16; j++) {
    int n = j * 256 + t;                 // ascending per-thread -> strict <
    float val = en2[n] - 2.0f * S[(size_t)b * 4096 + n];
    if (val < best) { best = val; bidx = n; }
  }
  rv[t] = best; ri[t] = bidx; __syncthreads();
  for (int off = 128; off > 0; off >>= 1) {
    if (t < off) {
      float v2 = rv[t + off]; int i2 = ri[t + off];
      if (v2 < rv[t] || (v2 == rv[t] && i2 < ri[t])) { rv[t] = v2; ri[t] = i2; }
    }
    __syncthreads();
  }
  const int idx = ri[0];
  __syncthreads();

  float ls = 0.0f;
  #pragma unroll
  for (int j = 0; j < 4; j++) {
    int e = j * 256 + t;
    float qv = E[(size_t)idx * 1024 + e];
    float xv = xenc[(size_t)b * 1024 + e];
    float d  = xv - qv;
    ls += d * d;
    out_q[(size_t)b * 1024 + e] = xv + (qv - xv);  // straight-through
  }
  rv[t] = ls; __syncthreads();
  for (int off = 128; off > 0; off >>= 1) {
    if (t < off) rv[t] += rv[t + off];
    __syncthreads();
  }
  if (t == 0) {
    out_idx[b] = (float)idx;
    atomicAdd(mse_accum, rv[0]);
  }
}

__global__ void finalize_kernel(const float* __restrict__ mse_accum,
                                float* __restrict__ out_losses)
{
  float mse = *mse_accum * (1.0f / (2048.0f * 1024.0f));
  out_losses[0] = 0.25f * mse * 0.1f;  // commitment_loss
  out_losses[1] = mse * 0.1f;          // codebook_loss
}

// ---------------------------------------------------------------------------
// Orchestration
// ---------------------------------------------------------------------------
extern "C" void kernel_launch(void* const* d_in, const int* in_sizes, int n_in,
                              void* d_out, int out_size, void* d_ws, size_t ws_size,
                              hipStream_t stream)
{
  (void)in_sizes; (void)n_in; (void)out_size; (void)ws_size;
  const float* cc = (const float*)d_in[0];
  const float* cr = (const float*)d_in[1];
  // d_in[2] seq_start_end: uniform segments of L=8, handled implicitly.
  const float* W1 = (const float*)d_in[3];  const float* b1 = (const float*)d_in[4];
  const float* W2 = (const float*)d_in[5];  const float* b2 = (const float*)d_in[6];
  const float* W3 = (const float*)d_in[7];  const float* b3 = (const float*)d_in[8];
  const float* Wq = (const float*)d_in[9];  const float* bq = (const float*)d_in[10];
  const float* Wk = (const float*)d_in[11]; const float* bk = (const float*)d_in[12];
  const float* Wv = (const float*)d_in[13]; const float* bv = (const float*)d_in[14];
  const float* Wm = (const float*)d_in[15]; const float* bm = (const float*)d_in[16];
  const float* E  = (const float*)d_in[17];

  char* ws = (char*)d_ws;
  const size_t MiB = 1024 * 1024;
  // Persistent bf16 weights
  unsigned short* W1t = (unsigned short*)(ws);             // [2048,4096] 16MiB
  unsigned short* W2t = (unsigned short*)(ws + 16 * MiB);  // [2048,2048]  8MiB
  unsigned short* W3t = (unsigned short*)(ws + 24 * MiB);  // [1024,2048]  4MiB
  unsigned short* Wqt = (unsigned short*)(ws + 28 * MiB);  // [1024,1024]  2MiB
  unsigned short* Wkt = (unsigned short*)(ws + 30 * MiB);
  unsigned short* Wvt = (unsigned short*)(ws + 32 * MiB);
  unsigned short* Wmt = (unsigned short*)(ws + 34 * MiB);
  unsigned short* Ebf = (unsigned short*)(ws + 36 * MiB);  // [4096,1024]  8MiB
  float* en2 = (float*)(ws + 44 * MiB);
  float* mse = (float*)(ws + 44 * MiB + 65536);
  // Ping-pong activation regions (64 MiB each)
  char* bufA = ws + 48 * MiB;
  char* bufB = ws + 112 * MiB;
  char* bufC = ws + 176 * MiB;
  unsigned short* ccb    = (unsigned short*)bufA;           // [16384,2048]
  unsigned short* crb    = (unsigned short*)bufB;           // [16384,2048]
  unsigned short* h1     = (unsigned short*)bufC;           // [16384,2048]
  unsigned short* h2     = (unsigned short*)bufA;           // [16384,2048]
  unsigned short* pe     = (unsigned short*)bufC;           // [16384,1024]
  unsigned short* qb     = (unsigned short*)bufA;           // [16384,1024]
  unsigned short* kb     = (unsigned short*)(bufA + 32 * MiB);
  unsigned short* vb     = (unsigned short*)bufB;           // [16384,1024]
  unsigned short* pooled = (unsigned short*)(bufB + 32 * MiB); // [2048,1024]
  float*          xenc   = (float*)(bufB + 40 * MiB);       // [2048,1024] fp32
  unsigned short* xencb  = (unsigned short*)(bufB + 48 * MiB);
  float*          Smat   = (float*)bufC;                    // [2048,4096] fp32

  float* out_q      = (float*)d_out;          // [2048*1024]
  float* out_losses = out_q + 2048 * 1024;    // commitment, codebook
  float* out_idx    = out_losses + 2;         // [2048]

  dim3 blk(256);
  // --- pre-pass: weight transpose-convert, activation/codebook converts ---
  transpose_convert<<<dim3(64, 128), blk, 0, stream>>>(W1, W1t, 4096, 2048);
  transpose_convert<<<dim3(64,  64), blk, 0, stream>>>(W2, W2t, 2048, 2048);
  transpose_convert<<<dim3(32,  64), blk, 0, stream>>>(W3, W3t, 2048, 1024);
  transpose_convert<<<dim3(32,  32), blk, 0, stream>>>(Wq, Wqt, 1024, 1024);
  transpose_convert<<<dim3(32,  32), blk, 0, stream>>>(Wk, Wkt, 1024, 1024);
  transpose_convert<<<dim3(32,  32), blk, 0, stream>>>(Wv, Wvt, 1024, 1024);
  transpose_convert<<<dim3(32,  32), blk, 0, stream>>>(Wm, Wmt, 1024, 1024);
  convert_bf16<<<131072, blk, 0, stream>>>(cc, ccb);
  convert_bf16<<<131072, blk, 0, stream>>>(cr, crb);
  convert_bf16<<<16384,  blk, 0, stream>>>(E, Ebf);
  enorm_kernel<<<4096,   blk, 0, stream>>>(E, en2);

  // --- MLP + projections (WMMA, bf16 chain) ---
  gemm_nt_bf16<1,1><<<dim3(16, 128), blk, 0, stream>>>(ccb, crb, 2048, W1t, b1, h1, 16384, 2048, 4096);
  gemm_nt_bf16<1,1><<<dim3(16, 128), blk, 0, stream>>>(h1, h1, 2048, W2t, b2, h2, 16384, 2048, 2048);
  gemm_nt_bf16<0,1><<<dim3(8, 128),  blk, 0, stream>>>(h2, h2, 2048, W3t, b3, pe, 16384, 1024, 2048);
  gemm_nt_bf16<0,1><<<dim3(8, 128),  blk, 0, stream>>>(pe, pe, 1024, Wqt, bq, qb, 16384, 1024, 1024);
  gemm_nt_bf16<0,1><<<dim3(8, 128),  blk, 0, stream>>>(pe, pe, 1024, Wkt, bk, kb, 16384, 1024, 1024);
  gemm_nt_bf16<0,1><<<dim3(8, 128),  blk, 0, stream>>>(pe, pe, 1024, Wvt, bv, vb, 16384, 1024, 1024);

  // --- attention + mean pool, head, VQ ---
  attn_pool<<<2048, dim3(64), 0, stream>>>(qb, kb, vb, pooled);
  gemm_nt_bf16<0,0><<<dim3(8, 16),  blk, 0, stream>>>(pooled, pooled, 1024, Wmt, bm, xenc, 2048, 1024, 1024);
  convert_bf16<<<8192, blk, 0, stream>>>(xenc, xencb);
  gemm_nt_bf16<0,0><<<dim3(32, 16), blk, 0, stream>>>(xencb, xencb, 1024, Ebf, nullptr, Smat, 2048, 4096, 1024);
  init_kernel<<<1, 1, 0, stream>>>(mse);
  vq_kernel<<<2048, blk, 0, stream>>>(Smat, en2, xenc, E, out_q, out_idx, mse);
  finalize_kernel<<<1, 1, 0, stream>>>(mse, out_losses);
}
